// CellFatePredictionHead_26963804684755
// MI455X (gfx1250) — compile-verified, matrix operand order checked by reading, not run on hardware
//
#include <hip/hip_runtime.h>
#include <hip/hip_bf16.h>

// ---------------- problem constants ----------------
#define NN 50000
#define EE 500000
#define DD 128

typedef __attribute__((ext_vector_type(8)))  float  v8f;
typedef __attribute__((ext_vector_type(8)))  __bf16 v8bf;
typedef __attribute__((ext_vector_type(16))) __bf16 v16bf;

// ---------------- helpers ----------------
__device__ __forceinline__ unsigned short f2bf(float f) {
    unsigned u = __float_as_uint(f);
    return (unsigned short)((u + 0x7FFFu + ((u >> 16) & 1u)) >> 16);  // RNE
}
__device__ __forceinline__ float gelu_exact(float x) {
    return 0.5f * x * (1.0f + erff(x * 0.70710678118654752f));
}
__device__ __forceinline__ unsigned fenc(float f) {  // order-preserving float->uint
    unsigned u = __float_as_uint(f);
    return (u >> 31) ? ~u : (u | 0x80000000u);
}
__device__ __forceinline__ float fdec(unsigned u) {
    u = (u >> 31) ? (u & 0x7FFFFFFFu) : ~u;
    return __uint_as_float(u);
}
// wave-internal LDS produce->consume ordering (DS ops are in-order per wave)
__device__ __forceinline__ void wsync() {
    asm volatile("s_wait_dscnt 0" ::: "memory");
}

__device__ __forceinline__ v8f wmma_bf16(v16bf a, v16bf b, v8f c) {
    return __builtin_amdgcn_wmma_f32_16x16x32_bf16(false, a, false, b, (short)0, c,
                                                   false, false);
}

// A-fragment (16x32 bf16, row-major tile in LDS, row stride 136 elems)
__device__ __forceinline__ v16bf ldsA(const unsigned short* tile, int kc, int lane) {
    int row = lane & 15;
    int kb  = kc * 32 + ((lane >> 4) << 3);   // +8 for lanes 16..31
    const unsigned short* p = tile + row * 136 + kb;
    union { v16bf v; v8bf h[2]; } u;
    u.h[0] = *(const v8bf*)p;
    u.h[1] = *(const v8bf*)(p + 16);
    return u.v;
}
// B-fragment pre-packed in global (1KB per fragment, 32B contiguous per lane)
__device__ __forceinline__ v16bf gblB(const unsigned short* wp, int frag, int lane) {
    return *(const v16bf*)(wp + (size_t)frag * 512 + lane * 16);
}

template <int NT>
__device__ __forceinline__ void zeroN(v8f* a) {
#pragma unroll
    for (int i = 0; i < NT; ++i)
#pragma unroll
        for (int j = 0; j < 8; ++j) a[i][j] = 0.0f;
}

template <int KC, int NT>
__device__ __forceinline__ void gemm_tile(const unsigned short* atile,
                                          const unsigned short* bp, int lane,
                                          v8f* acc) {
#pragma unroll
    for (int kc = 0; kc < KC; ++kc) {
        v16bf a = ldsA(atile, kc, lane);
#pragma unroll
        for (int nt = 0; nt < NT; ++nt) {
            v16bf b = gblB(bp, kc * NT + nt, lane);
            acc[nt]  = wmma_bf16(a, b, acc[nt]);
        }
    }
}

// ---------------- per-wave LDS ----------------
struct WaveLds {
    unsigned short tile[3][16][136];  // bf16 tiles, padded stride (16B aligned rows)
    float          red[32][8];
    float          red2[32][8];
    float          mu[16];
    float          rv[16];
    int            srcIdx[16];
    int            dstIdx[16];
    float          lrs[16];
    float          cf[16];
};

// ---------------- weight packing ----------------
__global__ void pack_w(const float* __restrict__ W, int rowOff, int NOr, int Kr,
                       int RS, int NT, int nfrag, unsigned short* __restrict__ dst) {
    int gid = blockIdx.x * 256 + threadIdx.x;
    if (gid >= nfrag * 512) return;
    int f = gid >> 9, w = gid & 511, lane = w >> 4, e = w & 15;
    int kc = f / NT, nt = f - kc * NT;
    int row = nt * 16 + (lane & 15);
    int kb  = kc * 32 + ((lane >> 4) << 3);
    int k   = (e < 8) ? (kb + e) : (kb + 16 + (e - 8));
    float v = (row < NOr && k < Kr) ? W[(size_t)(rowOff + row) * RS + k] : 0.0f;
    dst[gid] = f2bf(v);
}

__global__ void k_init(float* __restrict__ agg, float* __restrict__ denom,
                       unsigned* __restrict__ m_enc, float* __restrict__ w1last,
                       const float* __restrict__ w1) {
    size_t g = (size_t)blockIdx.x * 256 + threadIdx.x;
    if (g < (size_t)NN * 128) agg[g] = 0.0f;
    if (g < NN) { denom[g] = 0.0f; m_enc[g] = 0u; }
    if (g < 128) w1last[g] = w1[g * 257 + 256];
}

// ---------------- edge tile staging ----------------
__device__ __forceinline__ void edge_stage(WaveLds& w, int lane, int tile16,
                                           const float* __restrict__ z,
                                           const float* __restrict__ lr_scores,
                                           const int* __restrict__ ei) {
    if (lane < 16) {
        int e = tile16 * 16 + lane;
        int s_ = 0, d_ = 0; float l_ = 0.0f;
        if (e < EE) { s_ = ei[e]; d_ = ei[EE + e]; l_ = lr_scores[e]; }
        w.srcIdx[lane] = s_; w.dstIdx[lane] = d_; w.lrs[lane] = l_;
    }
    wsync();
#pragma unroll
    for (int t = 0; t < 2; ++t) {
        const int* rows = t ? w.dstIdx : w.srcIdx;
        for (int r = 0; r < 16; ++r) {
            int node  = rows[r];
            float4 v  = ((const float4*)(z + (size_t)node * DD))[lane];
            unsigned short* tp = &w.tile[t][r][lane * 4];
            tp[0] = f2bf(v.x); tp[1] = f2bf(v.y); tp[2] = f2bf(v.z); tp[3] = f2bf(v.w);
        }
    }
    wsync();
}

// lr_emb = gelu([zs|zd|lr]@W1^T+b1)@W2^T+b2  -> emb (f32 regs, bias included)
__device__ __forceinline__ void edge_gemm12(WaveLds& w, int lane,
                                            const unsigned short* __restrict__ w1p,
                                            const unsigned short* __restrict__ w2p,
                                            const float* __restrict__ w1last,
                                            const float* __restrict__ b1,
                                            const float* __restrict__ b2, v8f* emb) {
    int col = lane & 15, half = lane >> 4;
    v8f acc[8];
    zeroN<8>(acc);
    gemm_tile<4, 8>(&w.tile[0][0][0], w1p, lane, acc);                 // K 0..127 (zs)
    gemm_tile<4, 8>(&w.tile[1][0][0], w1p + 32 * 512, lane, acc);      // K 128..255 (zd)
#pragma unroll
    for (int nt = 0; nt < 8; ++nt) {
        int   Ncol = nt * 16 + col;
        float bv = b1[Ncol], wl = w1last[Ncol];
#pragma unroll
        for (int r = 0; r < 8; ++r) {
            int   m = r + half * 8;
            float x = acc[nt][r] + bv + w.lrs[m] * wl;   // rank-1: lr_score column
            w.tile[2][m][Ncol] = f2bf(gelu_exact(x));
        }
    }
    wsync();
    zeroN<8>(emb);
    gemm_tile<4, 8>(&w.tile[2][0][0], w2p, lane, emb);
#pragma unroll
    for (int nt = 0; nt < 8; ++nt) {
        float bv = b2[nt * 16 + col];
#pragma unroll
        for (int r = 0; r < 8; ++r) emb[nt][r] += bv;
    }
}

// ---------------- pass A: edge attention scores ----------------
__global__ __launch_bounds__(128) void k_edge_score(
    const float* __restrict__ z, const float* __restrict__ lr_scores,
    const int* __restrict__ ei, const unsigned short* __restrict__ w1p,
    const float* __restrict__ w1last, const float* __restrict__ b1,
    const unsigned short* __restrict__ w2p, const float* __restrict__ b2,
    const unsigned short* __restrict__ aw1p, const float* __restrict__ ab1,
    const float* __restrict__ aw2, const float* __restrict__ ab2,
    float* __restrict__ s_out) {
    __shared__ WaveLds lds[4];
    int lane = threadIdx.x & 31, wvid = threadIdx.x >> 5;
    int tile16 = blockIdx.x * 4 + wvid;
    if (tile16 >= EE / 16) return;
    WaveLds& w = lds[wvid];
    int col = lane & 15, half = lane >> 4;

    edge_stage(w, lane, tile16, z, lr_scores, ei);

    v8f emb[8];
    edge_gemm12(w, lane, w1p, w2p, w1last, b1, b2, emb);

    // stash lr_emb (bf16) into tile[0] (zs no longer needed)
#pragma unroll
    for (int nt = 0; nt < 8; ++nt)
#pragma unroll
        for (int r = 0; r < 8; ++r)
            w.tile[0][r + half * 8][nt * 16 + col] = f2bf(emb[nt][r]);
    wsync();

    // a = gelu([zd | lr_emb] @ attn_w1^T + attn_b1); s = a . attn_w2 + attn_b2
    v8f acc[8];
    zeroN<8>(acc);
    gemm_tile<4, 8>(&w.tile[1][0][0], aw1p, lane, acc);            // K 0..127 (zd)
    gemm_tile<4, 8>(&w.tile[0][0][0], aw1p + 32 * 512, lane, acc); // K 128..255 (emb)

    float p[8];
#pragma unroll
    for (int r = 0; r < 8; ++r) p[r] = 0.0f;
#pragma unroll
    for (int nt = 0; nt < 8; ++nt) {
        int   Ncol = nt * 16 + col;
        float abv = ab1[Ncol], a2 = aw2[Ncol];
#pragma unroll
        for (int r = 0; r < 8; ++r)
            p[r] += gelu_exact(acc[nt][r] + abv) * a2;
    }
#pragma unroll
    for (int r = 0; r < 8; ++r) w.red[lane][r] = p[r];
    wsync();
    if (lane < 16) {
        int   m = lane, base = (m < 8) ? 0 : 16, c = m & 7;
        float t = ab2[0];
#pragma unroll
        for (int j = 0; j < 16; ++j) t += w.red[base + j][c];
        int e = tile16 * 16 + m;
        if (e < EE) s_out[e] = t;
    }
}

// ---------------- pass B/C: edge-softmax stats ----------------
__global__ void k_max(const float* __restrict__ s, const int* __restrict__ ei,
                      unsigned* __restrict__ m_enc) {
    int e = blockIdx.x * 256 + threadIdx.x;
    if (e < EE) atomicMax(&m_enc[ei[EE + e]], fenc(s[e]));
}
__global__ void k_sum(const float* __restrict__ s, const int* __restrict__ ei,
                      const unsigned* __restrict__ m_enc, float* __restrict__ denom) {
    int e = blockIdx.x * 256 + threadIdx.x;
    if (e < EE) {
        int d = ei[EE + e];
        atomicAdd(&denom[d], expf(s[e] - fdec(m_enc[d])));
    }
}

// ---------------- pass D: weighted scatter of lr_emb ----------------
__global__ __launch_bounds__(128) void k_edge_scatter(
    const float* __restrict__ z, const float* __restrict__ lr_scores,
    const int* __restrict__ ei, const unsigned short* __restrict__ w1p,
    const float* __restrict__ w1last, const float* __restrict__ b1,
    const unsigned short* __restrict__ w2p, const float* __restrict__ b2,
    const float* __restrict__ s, const unsigned* __restrict__ m_enc,
    const float* __restrict__ denom, float* __restrict__ agg) {
    __shared__ WaveLds lds[4];
    int lane = threadIdx.x & 31, wvid = threadIdx.x >> 5;
    int tile16 = blockIdx.x * 4 + wvid;
    if (tile16 >= EE / 16) return;
    WaveLds& w = lds[wvid];
    int col = lane & 15, half = lane >> 4;

    edge_stage(w, lane, tile16, z, lr_scores, ei);

    if (lane < 16) {
        int   e = tile16 * 16 + lane;
        float c = 0.0f;
        if (e < EE) {
            int   d = w.dstIdx[lane];
            float c_e = expf(s[e] - fdec(m_enc[d]));
            c = c_e / denom[d] * w.lrs[lane];
        }
        w.cf[lane] = c;
    }

    v8f emb[8];
    edge_gemm12(w, lane, w1p, w2p, w1last, b1, b2, emb);
    wsync();

#pragma unroll
    for (int nt = 0; nt < 8; ++nt) {
        int Ncol = nt * 16 + col;
#pragma unroll
        for (int r = 0; r < 8; ++r) {
            int m = r + half * 8;
            atomicAdd(&agg[(size_t)w.dstIdx[m] * DD + Ncol], w.cf[m] * emb[nt][r]);
        }
    }
}

// ---------------- pass E: node heads ----------------
__global__ __launch_bounds__(128) void k_nodes(
    const float* __restrict__ agg, const unsigned short* __restrict__ wvp,
    const float* __restrict__ bv, const unsigned short* __restrict__ wop,
    const float* __restrict__ bo, const unsigned short* __restrict__ c1p,
    const float* __restrict__ cb1, const float* __restrict__ lng,
    const float* __restrict__ lnb, const unsigned short* __restrict__ c2p,
    const float* __restrict__ cb2, const unsigned short* __restrict__ t1p,
    const float* __restrict__ tb1, const unsigned short* __restrict__ t2p,
    const float* __restrict__ tb2, const unsigned short* __restrict__ d1p,
    const float* __restrict__ db1, const float* __restrict__ dw2,
    const float* __restrict__ db2, float* __restrict__ out_fl,
    float* __restrict__ out_traj, float* __restrict__ out_diff,
    float* __restrict__ out_frep) {
    __shared__ WaveLds lds[4];
    int lane = threadIdx.x & 31, wvid = threadIdx.x >> 5;
    int tile16 = blockIdx.x * 4 + wvid;
    if (tile16 >= NN / 16) return;
    WaveLds& w = lds[wvid];
    int col = lane & 15, half = lane >> 4;

    // stage agg tile (bf16) into tile[0]
    for (int r = 0; r < 16; ++r) {
        int    node = tile16 * 16 + r;
        float4 v = ((const float4*)(agg + (size_t)node * DD))[lane];
        unsigned short* tp = &w.tile[0][r][lane * 4];
        tp[0] = f2bf(v.x); tp[1] = f2bf(v.y); tp[2] = f2bf(v.z); tp[3] = f2bf(v.w);
    }
    wsync();

    v8f acc[8];
    // vh = agg @ wv^T + bv  -> tile[1]
    zeroN<8>(acc);
    gemm_tile<4, 8>(&w.tile[0][0][0], wvp, lane, acc);
#pragma unroll
    for (int nt = 0; nt < 8; ++nt) {
        int   Ncol = nt * 16 + col;
        float b = bv[Ncol];
#pragma unroll
        for (int r = 0; r < 8; ++r)
            w.tile[1][r + half * 8][Ncol] = f2bf(acc[nt][r] + b);
    }
    wsync();

    // fate_rep = vh @ mha_out_w^T + bo  -> out_frep (f32) and tile[0] (bf16)
    zeroN<8>(acc);
    gemm_tile<4, 8>(&w.tile[1][0][0], wop, lane, acc);
#pragma unroll
    for (int nt = 0; nt < 8; ++nt) {
        int   Ncol = nt * 16 + col;
        float b = bo[Ncol];
#pragma unroll
        for (int r = 0; r < 8; ++r) {
            int   m = r + half * 8;
            float x = acc[nt][r] + b;
            out_frep[(size_t)(tile16 * 16 + m) * DD + Ncol] = x;
            w.tile[0][m][Ncol] = f2bf(x);
        }
    }
    wsync();

    // ---- classifier head: h = frep@clf_w1^T+cb1 ; LN ; gelu ; @clf_w2^T+cb2 ----
    v8f hx[8];
    zeroN<8>(hx);
    gemm_tile<4, 8>(&w.tile[0][0][0], c1p, lane, hx);
    float sm_[8], sq_[8];
#pragma unroll
    for (int r = 0; r < 8; ++r) { sm_[r] = 0.0f; sq_[r] = 0.0f; }
#pragma unroll
    for (int nt = 0; nt < 8; ++nt) {
        float b = cb1[nt * 16 + col];
#pragma unroll
        for (int r = 0; r < 8; ++r) {
            float x = hx[nt][r] + b;
            hx[nt][r] = x;
            sm_[r] += x;
            sq_[r] += x * x;
        }
    }
#pragma unroll
    for (int r = 0; r < 8; ++r) { w.red[lane][r] = sm_[r]; w.red2[lane][r] = sq_[r]; }
    wsync();
    if (lane < 16) {
        int   m = lane, base = (m < 8) ? 0 : 16, c = m & 7;
        float S = 0.0f, Q = 0.0f;
#pragma unroll
        for (int j = 0; j < 16; ++j) { S += w.red[base + j][c]; Q += w.red2[base + j][c]; }
        float mu  = S * (1.0f / 128.0f);
        float var = Q * (1.0f / 128.0f) - mu * mu;
        w.mu[m] = mu;
        w.rv[m] = rsqrtf(var + 1e-5f);
    }
    wsync();
#pragma unroll
    for (int nt = 0; nt < 8; ++nt) {
        int   Ncol = nt * 16 + col;
        float g = lng[Ncol], b = lnb[Ncol];
#pragma unroll
        for (int r = 0; r < 8; ++r) {
            int   m = r + half * 8;
            float xn = (hx[nt][r] - w.mu[m]) * w.rv[m] * g + b;
            w.tile[1][m][Ncol] = f2bf(gelu_exact(xn));
        }
    }
    wsync();
    {
        v8f a1;
        zeroN<1>(&a1);
        gemm_tile<4, 1>(&w.tile[1][0][0], c2p, lane, &a1);
        if (col < 10) {
            float b = cb2[col];
#pragma unroll
            for (int r = 0; r < 8; ++r) {
                int node = tile16 * 16 + r + half * 8;
                out_fl[(size_t)node * 10 + col] = a1[r] + b;
            }
        }
    }

    // ---- trajectory head ----
    zeroN<8>(acc);
    gemm_tile<4, 8>(&w.tile[0][0][0], t1p, lane, acc);
#pragma unroll
    for (int nt = 0; nt < 8; ++nt) {
        int   Ncol = nt * 16 + col;
        float b = tb1[Ncol];
#pragma unroll
        for (int r = 0; r < 8; ++r)
            w.tile[1][r + half * 8][Ncol] = f2bf(gelu_exact(acc[nt][r] + b));
    }
    wsync();
    {
        v8f a2[4];
        zeroN<4>(a2);
        gemm_tile<4, 4>(&w.tile[1][0][0], t2p, lane, a2);
#pragma unroll
        for (int nt = 0; nt < 4; ++nt) {
            int   Ncol = nt * 16 + col;
            float b = tb2[Ncol];
#pragma unroll
            for (int r = 0; r < 8; ++r) {
                int node = tile16 * 16 + r + half * 8;
                out_traj[(size_t)node * 64 + Ncol] = a2[nt][r] + b;
            }
        }
    }

    // ---- differentiation head ----
    {
        v8f a3[4];
        zeroN<4>(a3);
        gemm_tile<4, 4>(&w.tile[0][0][0], d1p, lane, a3);
        float p[8];
#pragma unroll
        for (int r = 0; r < 8; ++r) p[r] = 0.0f;
#pragma unroll
        for (int nt = 0; nt < 4; ++nt) {
            int   Ncol = nt * 16 + col;
            float b = db1[Ncol], dv = dw2[Ncol];
#pragma unroll
            for (int r = 0; r < 8; ++r)
                p[r] += gelu_exact(a3[nt][r] + b) * dv;
        }
#pragma unroll
        for (int r = 0; r < 8; ++r) w.red[lane][r] = p[r];
        wsync();
        if (lane < 16) {
            int   m = lane, base = (m < 8) ? 0 : 16, c = m & 7;
            float t = db2[0];
#pragma unroll
            for (int j = 0; j < 16; ++j) t += w.red[base + j][c];
            out_diff[tile16 * 16 + m] = 1.0f / (1.0f + expf(-t));
        }
    }
}

// ---------------- host ----------------
extern "C" void kernel_launch(void* const* d_in, const int* in_sizes, int n_in,
                              void* d_out, int out_size, void* d_ws, size_t ws_size,
                              hipStream_t stream) {
    const float* z         = (const float*)d_in[0];
    const float* lr_scores = (const float*)d_in[1];
    const float* w1        = (const float*)d_in[2];
    const float* b1        = (const float*)d_in[3];
    const float* w2        = (const float*)d_in[4];
    const float* b2        = (const float*)d_in[5];
    const float* aw1       = (const float*)d_in[6];
    const float* ab1       = (const float*)d_in[7];
    const float* aw2       = (const float*)d_in[8];
    const float* ab2       = (const float*)d_in[9];
    const float* mha_in_w  = (const float*)d_in[11];
    const float* mha_in_b  = (const float*)d_in[12];
    const float* mha_out_w = (const float*)d_in[13];
    const float* mha_out_b = (const float*)d_in[14];
    const float* clf_w1    = (const float*)d_in[15];
    const float* clf_b1    = (const float*)d_in[16];
    const float* ln_g      = (const float*)d_in[17];
    const float* ln_b      = (const float*)d_in[18];
    const float* clf_w2    = (const float*)d_in[19];
    const float* clf_b2    = (const float*)d_in[20];
    const float* traj_w1   = (const float*)d_in[21];
    const float* traj_b1   = (const float*)d_in[22];
    const float* traj_w2   = (const float*)d_in[23];
    const float* traj_b2   = (const float*)d_in[24];
    const float* diff_w1   = (const float*)d_in[25];
    const float* diff_b1   = (const float*)d_in[26];
    const float* diff_w2   = (const float*)d_in[27];
    const float* diff_b2   = (const float*)d_in[28];
    const int*   ei        = (const int*)d_in[29];

    char* ws = (char*)d_ws;
    unsigned short* w1p   = (unsigned short*)(ws + 0);        // 64KB  (KC8,NT8)
    unsigned short* w2p   = (unsigned short*)(ws + 65536);    // 32KB  (KC4,NT8)
    unsigned short* aw1p  = (unsigned short*)(ws + 98304);    // 64KB  (KC8,NT8)
    unsigned short* wvp   = (unsigned short*)(ws + 163840);   // 32KB
    unsigned short* wop   = (unsigned short*)(ws + 196608);   // 32KB
    unsigned short* c1p   = (unsigned short*)(ws + 229376);   // 32KB
    unsigned short* c2p   = (unsigned short*)(ws + 262144);   // 4KB   (KC4,NT1)
    unsigned short* t1p   = (unsigned short*)(ws + 266240);   // 32KB
    unsigned short* t2p   = (unsigned short*)(ws + 299008);   // 16KB  (KC4,NT4)
    unsigned short* d1p   = (unsigned short*)(ws + 315392);   // 16KB  (KC4,NT4)
    float*          w1last = (float*)(ws + 331776);           // 512B
    float*          s_buf  = (float*)(ws + 332800);           // E*4
    unsigned*       m_enc  = (unsigned*)(ws + 2332800);       // N*4
    float*          denom  = (float*)(ws + 2532800);          // N*4
    float*          agg    = (float*)(ws + 2732800);          // N*128*4

    float* out     = (float*)d_out;
    float* out_fl  = out;                  // [N,10]
    float* out_tr  = out + (size_t)NN * 10;  // [N,64]
    float* out_df  = out + (size_t)NN * 74;  // [N,1]
    float* out_fr  = out + (size_t)NN * 75;  // [N,128]

    auto packBlocks = [](int nfrag) { return (nfrag * 512 + 255) / 256; };
    // (W, rowOff, NOr, Kr, RS, NT, nfrag, dst)
    pack_w<<<packBlocks(64), 256, 0, stream>>>(w1, 0, 128, 256, 257, 8, 64, w1p);
    pack_w<<<packBlocks(32), 256, 0, stream>>>(w2, 0, 128, 128, 128, 8, 32, w2p);
    pack_w<<<packBlocks(64), 256, 0, stream>>>(aw1, 0, 128, 256, 256, 8, 64, aw1p);
    pack_w<<<packBlocks(32), 256, 0, stream>>>(mha_in_w, 256, 128, 128, 128, 8, 32, wvp);
    pack_w<<<packBlocks(32), 256, 0, stream>>>(mha_out_w, 0, 128, 128, 128, 8, 32, wop);
    pack_w<<<packBlocks(32), 256, 0, stream>>>(clf_w1, 0, 128, 128, 128, 8, 32, c1p);
    pack_w<<<packBlocks(4),  256, 0, stream>>>(clf_w2, 0, 10, 128, 128, 1, 4, c2p);
    pack_w<<<packBlocks(32), 256, 0, stream>>>(traj_w1, 0, 128, 128, 128, 8, 32, t1p);
    pack_w<<<packBlocks(16), 256, 0, stream>>>(traj_w2, 0, 64, 128, 128, 4, 16, t2p);
    pack_w<<<packBlocks(16), 256, 0, stream>>>(diff_w1, 0, 64, 128, 128, 4, 16, d1p);

    k_init<<<((size_t)NN * 128 + 255) / 256, 256, 0, stream>>>(agg, denom, m_enc,
                                                               w1last, w1);

    int edgeTiles = EE / 16;                 // 31250
    int edgeBlks  = (edgeTiles + 3) / 4;     // 7813
    k_edge_score<<<edgeBlks, 128, 0, stream>>>(z, lr_scores, ei, w1p, w1last, b1, w2p,
                                               b2, aw1p, ab1, aw2, ab2, s_buf);

    int eBlk = (EE + 255) / 256;
    k_max<<<eBlk, 256, 0, stream>>>(s_buf, ei, m_enc);
    k_sum<<<eBlk, 256, 0, stream>>>(s_buf, ei, m_enc, denom);

    k_edge_scatter<<<edgeBlks, 128, 0, stream>>>(z, lr_scores, ei, w1p, w1last, b1,
                                                 w2p, b2, s_buf, m_enc, denom, agg);

    int nodeTiles = NN / 16;                 // 3125
    int nodeBlks  = (nodeTiles + 3) / 4;     // 782
    k_nodes<<<nodeBlks, 128, 0, stream>>>(
        agg, wvp, mha_in_b + 256, wop, mha_out_b, c1p, clf_b1, ln_g, ln_b, c2p,
        clf_b2, t1p, traj_b1, t2p, traj_b2, d1p, diff_b1, diff_w2, diff_b2, out_fl,
        out_tr, out_df, out_fr);
}
